// GATModel_13108240187441
// MI455X (gfx1250) — matrix operand order repaired and checked by reading, not compiled, not run on hardware
//
#include <hip/hip_runtime.h>
#include <hip/hip_bf16.h>

#define NN 50000               // nodes
#define NE 800000              // edges (before self loops)
#define ET (NE + NN)           // edges incl self loops
#define NG 64                  // graphs
#define F  128                 // feature width at every layer boundary
#define MT (NN / 16)           // 3125 row tiles (exact)

typedef __attribute__((ext_vector_type(16))) __bf16 v16bf;
typedef __attribute__((ext_vector_type(8)))  float  v8f;

// ---- order-preserving float <-> uint map for atomic max ----
__device__ __forceinline__ unsigned encf(float f) {
  unsigned u = __float_as_uint(f);
  return (u & 0x80000000u) ? ~u : (u | 0x80000000u);
}
__device__ __forceinline__ float decf(unsigned u) {
  return __uint_as_float((u & 0x80000000u) ? (u & 0x7fffffffu) : ~u);
}

__device__ __forceinline__ void edge_sd(const int* __restrict__ ei, int e, int& s, int& d) {
  if (e < NE) { s = ei[e]; d = ei[NE + e]; }
  else        { s = e - NE; d = e - NE; }           // appended self loops
}

// ---- pre-fragment W (128x128 fp32) into bf16 hi/lo WMMA B fragments ----
// frag index = ((nt*4 + kk)*32 + lane)*16 + j ; B[k][n], k = kk*32 + (lane>>4)*8 + (j&7) + (j>>3)*16,
// n = (lane&15) + nt*16  (per CDNA5 16-bit B 32x16 layout)
__global__ void k_wfrag(const float* __restrict__ W, __bf16* __restrict__ whi,
                        __bf16* __restrict__ wlo) {
  int t = blockIdx.x * blockDim.x + threadIdx.x;
  if (t >= 8 * 4 * 32 * 16) return;
  int j = t & 15, lane = (t >> 4) & 31, kk = (t >> 9) & 3, nt = t >> 11;
  int k = kk * 32 + ((lane >> 4) << 3) + (j & 7) + ((j >> 3) << 4);
  int n = (lane & 15) + nt * 16;
  float f = W[k * F + n];
  __bf16 hv = (__bf16)f;
  whi[t] = hv;
  wlo[t] = (__bf16)(f - (float)hv);
}

// ---- xl = h @ W via 3x bf16-split WMMA (fp32-accurate). One wave = 16x128 strip ----
__global__ void k_gemm(const float* __restrict__ hin, const __bf16* __restrict__ whi,
                       const __bf16* __restrict__ wlo, float* __restrict__ xl) {
  int wid = blockIdx.x * (blockDim.x >> 5) + (threadIdx.x >> 5);
  if (wid >= MT) return;                       // wave-uniform guard (EXEC all-1 for WMMA)
  int lane = threadIdx.x & 31;
  int m = lane & 15, sel = lane >> 4;
  const float* arow = hin + (size_t)(wid * 16 + m) * F;
  const v16bf* BH = (const v16bf*)whi;
  const v16bf* BL = (const v16bf*)wlo;
  v8f acc[8];
  v8f zero = {0.f, 0.f, 0.f, 0.f, 0.f, 0.f, 0.f, 0.f};
#pragma unroll
  for (int nt = 0; nt < 8; ++nt) acc[nt] = zero;

#pragma unroll
  for (int kk = 0; kk < 4; ++kk) {
    // A fragment (16-bit A 16x32 layout): elems 0..7 -> K = kk*32+sel*8+j ;
    // elems 8..15 -> K = kk*32+sel*8+16+j
    const float* ap = arow + kk * 32 + sel * 8;
    float af[16];
#pragma unroll
    for (int j = 0; j < 8; ++j) { af[j] = ap[j]; af[8 + j] = ap[16 + j]; }
    v16bf ah, al;
#pragma unroll
    for (int j = 0; j < 16; ++j) {
      __bf16 hv = (__bf16)af[j];
      ah[j] = hv;
      al[j] = (__bf16)(af[j] - (float)hv);
    }
#pragma unroll
    for (int nt = 0; nt < 8; ++nt) {
      v16bf bh = BH[(nt * 4 + kk) * 32 + lane];
      v16bf bl = BL[(nt * 4 + kk) * 32 + lane];
      acc[nt] = __builtin_amdgcn_wmma_f32_16x16x32_bf16(false, ah, false, bh,
                                                        (short)0, acc[nt], false, false);
      acc[nt] = __builtin_amdgcn_wmma_f32_16x16x32_bf16(false, ah, false, bl,
                                                        (short)0, acc[nt], false, false);
      acc[nt] = __builtin_amdgcn_wmma_f32_16x16x32_bf16(false, al, false, bh,
                                                        (short)0, acc[nt], false, false);
    }
  }
  // C/D layout: row = 16*wid + sel*8 + r, col = nt*16 + (lane&15)
  int col = lane & 15;
#pragma unroll
  for (int nt = 0; nt < 8; ++nt)
#pragma unroll
    for (int r = 0; r < 8; ++r)
      xl[(size_t)(wid * 16 + sel * 8 + r) * F + nt * 16 + col] = acc[nt][r];
}

// ---- per-node attention dots: nsrc[n,h] = <xl[n,h,:], a_s[h,:]>, same for ndst ----
template <int H, int C>
__global__ void k_dots(const float* __restrict__ xl, const float* __restrict__ a_s,
                       const float* __restrict__ a_d, float* __restrict__ nsrc,
                       float* __restrict__ ndst) {
  int t = blockIdx.x * blockDim.x + threadIdx.x;
  if (t >= NN * H) return;
  int n = t / H, h = t % H;
  const float* row = xl + (size_t)n * F + h * C;
  float s0 = 0.f, s1 = 0.f;
#pragma unroll 8
  for (int c = 0; c < C; ++c) {
    float v = row[c];
    s0 += v * a_s[h * C + c];
    s1 += v * a_d[h * C + c];
  }
  nsrc[t] = s0;
  ndst[t] = s1;
}

// ---- init out=bias, m_enc=0 (== -inf sentinel, always overwritten via self loop), ssum=0 ----
template <int H>
__global__ void k_prep(float* __restrict__ hout, const float* __restrict__ bias,
                       unsigned* __restrict__ menc, float* __restrict__ ssum) {
  int t = blockIdx.x * blockDim.x + threadIdx.x;
  if (t >= NN * F) return;
  hout[t] = bias[t & (F - 1)];
  if (t < NN * H) { menc[t] = 0u; ssum[t] = 0.f; }
}

template <int H>
__global__ void k_emax(const int* __restrict__ ei, const float* __restrict__ nsrc,
                       const float* __restrict__ ndst, unsigned* __restrict__ menc) {
  int t = blockIdx.x * blockDim.x + threadIdx.x;
  if (t >= ET * H) return;
  int e = t / H, h = t % H, s, d;
  edge_sd(ei, e, s, d);
  float a = nsrc[s * H + h] + ndst[d * H + h];
  a = a > 0.f ? a : 0.2f * a;                    // leaky_relu 0.2
  atomicMax(menc + d * H + h, encf(a));
}

template <int H>
__global__ void k_esum(const int* __restrict__ ei, const float* __restrict__ nsrc,
                       const float* __restrict__ ndst, const unsigned* __restrict__ menc,
                       float* __restrict__ ssum) {
  int t = blockIdx.x * blockDim.x + threadIdx.x;
  if (t >= ET * H) return;
  int e = t / H, h = t % H, s, d;
  edge_sd(ei, e, s, d);
  float a = nsrc[s * H + h] + ndst[d * H + h];
  a = a > 0.f ? a : 0.2f * a;
  atomicAdd(ssum + d * H + h, expf(a - decf(menc[d * H + h])));
}

// ---- one wave per edge: out[d,:] += w[h] * xl[s,:]  (4 floats / lane) ----
template <int H, int C>
__global__ void k_msg(const int* __restrict__ ei, const float* __restrict__ xl,
                      const float* __restrict__ nsrc, const float* __restrict__ ndst,
                      const unsigned* __restrict__ menc, const float* __restrict__ ssum,
                      float* __restrict__ out) {
  int wid = blockIdx.x * (blockDim.x >> 5) + (threadIdx.x >> 5);
  if (wid >= ET) return;
  int lane = threadIdx.x & 31;
  int s, d;
  edge_sd(ei, wid, s, d);
  int h = (lane * 4) / C;                        // 4-float strip stays within a head
  float a = nsrc[s * H + h] + ndst[d * H + h];
  a = a > 0.f ? a : 0.2f * a;
  float w = expf(a - decf(menc[d * H + h])) / (ssum[d * H + h] + 1e-16f);
  const float4 xv = *(const float4*)(xl + (size_t)s * F + lane * 4);
  float* op = out + (size_t)d * F + lane * 4;
  atomicAdd(op + 0, xv.x * w);
  atomicAdd(op + 1, xv.y * w);
  atomicAdd(op + 2, xv.z * w);
  atomicAdd(op + 3, xv.w * w);
}

__global__ void k_elu(float* __restrict__ h) {
  int t = blockIdx.x * blockDim.x + threadIdx.x;
  if (t >= NN * F) return;
  float x = h[t];
  h[t] = x > 0.f ? x : expm1f(x);
}

// ---- pooling ----
__global__ void k_poolinit(float* __restrict__ counts, float* __restrict__ mean,
                           unsigned* __restrict__ maxenc) {
  int t = blockIdx.x * blockDim.x + threadIdx.x;
  if (t < NG) counts[t] = 0.f;
  if (t < NG * F) { mean[t] = 0.f; maxenc[t] = 0u; }
}
__global__ void k_counts(const int* __restrict__ batch, float* __restrict__ counts) {
  int n = blockIdx.x * blockDim.x + threadIdx.x;
  if (n >= NN) return;
  atomicAdd(counts + batch[n], 1.0f);
}
__global__ void k_pool(const float* __restrict__ h, const int* __restrict__ batch,
                       float* __restrict__ mean, unsigned* __restrict__ maxenc) {
  int t = blockIdx.x * blockDim.x + threadIdx.x;
  if (t >= NN * F) return;
  int n = t >> 7, j = t & (F - 1);
  int g = batch[n];
  float v = h[t];
  atomicAdd(mean + g * F + j, v);
  atomicMax(maxenc + g * F + j, encf(v));
}

// ---- classifier: g=[mean,max] [64,256] -> relu(g@cw1+cb1) @ cw2 + cb2 ----
__global__ void k_cls(const float* __restrict__ mean, const unsigned* __restrict__ maxenc,
                      const float* __restrict__ counts, const float* __restrict__ cw1,
                      const float* __restrict__ cb1, const float* __restrict__ cw2,
                      const float* __restrict__ cb2, float* __restrict__ out) {
  __shared__ float gv[256];
  __shared__ float red[128];
  int g = blockIdx.x, j = threadIdx.x;
  float cnt = fmaxf(counts[g], 1.0f);
  gv[j] = mean[g * F + j] / cnt;
  unsigned u = maxenc[g * F + j];
  gv[128 + j] = (u == 0u) ? 0.0f : decf(u);      // empty-graph -> 0 (isfinite guard)
  __syncthreads();
  float acc = cb1[j];
  for (int k = 0; k < 256; ++k) acc += gv[k] * cw1[k * 128 + j];
  acc = fmaxf(acc, 0.f);
  red[j] = acc * cw2[j];
  __syncthreads();
  for (int off = 64; off > 0; off >>= 1) {
    if (j < off) red[j] += red[j + off];
    __syncthreads();
  }
  if (j == 0) out[g] = red[0] + cb2[0];
}

// ================= host =================
template <int H, int C>
static void run_layer(const float* hin, const float* Wl, const float* asl, const float* adl,
                      const float* bl, const int* ei, float* xl, float* hout, float* nsrc,
                      float* ndst, unsigned* menc, float* ssum, __bf16* whi, __bf16* wlo,
                      hipStream_t stream) {
  k_wfrag<<<64, 256, 0, stream>>>(Wl, whi, wlo);
  k_gemm<<<(MT + 7) / 8, 256, 0, stream>>>(hin, whi, wlo, xl);
  k_dots<H, C><<<(NN * H + 255) / 256, 256, 0, stream>>>(xl, asl, adl, nsrc, ndst);
  k_prep<H><<<(NN * F + 255) / 256, 256, 0, stream>>>(hout, bl, menc, ssum);
  k_emax<H><<<(ET * H + 255) / 256, 256, 0, stream>>>(ei, nsrc, ndst, menc);
  k_esum<H><<<(ET * H + 255) / 256, 256, 0, stream>>>(ei, nsrc, ndst, menc, ssum);
  k_msg<H, C><<<(ET + 7) / 8, 256, 0, stream>>>(ei, xl, nsrc, ndst, menc, ssum, hout);
}

extern "C" void kernel_launch(void* const* d_in, const int* in_sizes, int n_in,
                              void* d_out, int out_size, void* d_ws, size_t ws_size,
                              hipStream_t stream) {
  const float* x     = (const float*)d_in[0];
  const int*   ei    = (const int*)d_in[1];
  const int*   batch = (const int*)d_in[2];
  const float* W[4]  = {(const float*)d_in[3], (const float*)d_in[7],
                        (const float*)d_in[11], (const float*)d_in[15]};
  const float* AS[4] = {(const float*)d_in[4], (const float*)d_in[8],
                        (const float*)d_in[12], (const float*)d_in[16]};
  const float* AD[4] = {(const float*)d_in[5], (const float*)d_in[9],
                        (const float*)d_in[13], (const float*)d_in[17]};
  const float* B[4]  = {(const float*)d_in[6], (const float*)d_in[10],
                        (const float*)d_in[14], (const float*)d_in[18]};
  const float* cw1 = (const float*)d_in[19];
  const float* cb1 = (const float*)d_in[20];
  const float* cw2 = (const float*)d_in[21];
  const float* cb2 = (const float*)d_in[22];

  char* w = (char*)d_ws;
  auto take = [&](size_t bytes) { char* p = w; w += (bytes + 255) & ~size_t(255); return p; };
  float*    xl     = (float*)take((size_t)NN * F * 4);
  float*    hbuf   = (float*)take((size_t)NN * F * 4);
  float*    nsrc   = (float*)take((size_t)NN * 4 * 4);
  float*    ndst   = (float*)take((size_t)NN * 4 * 4);
  unsigned* menc   = (unsigned*)take((size_t)NN * 4 * 4);
  float*    ssum   = (float*)take((size_t)NN * 4 * 4);
  __bf16*   whi    = (__bf16*)take(16384 * 2);
  __bf16*   wlo    = (__bf16*)take(16384 * 2);
  float*    counts = (float*)take(NG * 4);
  float*    meanb  = (float*)take((size_t)NG * F * 4);
  unsigned* maxenc = (unsigned*)take((size_t)NG * F * 4);

  // layer 0 (reads input x), then layers 1-3 in-place ping on hbuf (dead after GEMM+dots)
  run_layer<4, 32>(x, W[0], AS[0], AD[0], B[0], ei, xl, hbuf, nsrc, ndst, menc, ssum, whi, wlo, stream);
  k_elu<<<(NN * F + 255) / 256, 256, 0, stream>>>(hbuf);
  run_layer<4, 32>(hbuf, W[1], AS[1], AD[1], B[1], ei, xl, hbuf, nsrc, ndst, menc, ssum, whi, wlo, stream);
  k_elu<<<(NN * F + 255) / 256, 256, 0, stream>>>(hbuf);
  run_layer<4, 32>(hbuf, W[2], AS[2], AD[2], B[2], ei, xl, hbuf, nsrc, ndst, menc, ssum, whi, wlo, stream);
  k_elu<<<(NN * F + 255) / 256, 256, 0, stream>>>(hbuf);
  run_layer<1, 128>(hbuf, W[3], AS[3], AD[3], B[3], ei, xl, hbuf, nsrc, ndst, menc, ssum, whi, wlo, stream);

  k_poolinit<<<(NG * F + 255) / 256, 256, 0, stream>>>(counts, meanb, maxenc);
  k_counts<<<(NN + 255) / 256, 256, 0, stream>>>(batch, counts);
  k_pool<<<(NN * F + 255) / 256, 256, 0, stream>>>(hbuf, batch, meanb, maxenc);
  k_cls<<<NG, 128, 0, stream>>>(meanb, maxenc, counts, cw1, cb1, cw2, cb2, (float*)d_out);
}
